// SwinTransformerBlock3D_29867202576464
// MI455X (gfx1250) — compile-verified, matrix operand order checked by reading, not run on hardware
//
#include <hip/hip_runtime.h>
#include <hip/hip_bf16.h>

typedef __attribute__((ext_vector_type(16))) _Float16 v16h;
typedef __attribute__((ext_vector_type(8)))  _Float16 v8h;
typedef __attribute__((ext_vector_type(2)))  _Float16 v2h;
typedef __attribute__((ext_vector_type(8)))  float    v8f;
typedef __attribute__((ext_vector_type(4)))  unsigned u4v;
typedef __attribute__((ext_vector_type(8)))  unsigned u8v;

__device__ __forceinline__ v16h load_frag(const _Float16* p) {
    v8h lo = *(const v8h*)(p);
    v8h hi = *(const v8h*)(p + 16);
    v16h r;
#pragma unroll
    for (int e = 0; e < 8; ++e) { r[e] = lo[e]; r[e + 8] = hi[e]; }
    return r;
}

// ---------------------------------------------------------------------------
// Weight convert: src f32 [K][N] row-major -> dst f16 [N][K] (transposed)
// ---------------------------------------------------------------------------
__global__ void wcvt_kernel(const float* __restrict__ src, _Float16* __restrict__ dst,
                            int K, int N) {
    int gid = blockIdx.x * blockDim.x + threadIdx.x;
    if (gid >= K * N) return;
    int n = gid % N, k = gid / N;
    dst[(size_t)n * K + k] = (_Float16)src[(size_t)k * N + n];
}

// ---------------------------------------------------------------------------
// LayerNorm over C=192, one wave32 per token, 8 waves per block. f32 in, f16 out
// ---------------------------------------------------------------------------
__global__ void ln_kernel(const float* __restrict__ x, const float* __restrict__ g,
                          const float* __restrict__ b, _Float16* __restrict__ out,
                          long ntok) {
    const int lane = threadIdx.x & 31;
    const int wave = threadIdx.x >> 5;
    const long tok = (long)blockIdx.x * 8 + wave;
    if (tok >= ntok) return;
    const float* px = x + tok * 192;
    float v[6];
    float s = 0.f;
#pragma unroll
    for (int j = 0; j < 6; ++j) { v[j] = px[lane + j * 32]; s += v[j]; }
#pragma unroll
    for (int off = 16; off > 0; off >>= 1) s += __shfl_xor(s, off, 32);
    const float mean = s * (1.f / 192.f);
    float vs = 0.f;
#pragma unroll
    for (int j = 0; j < 6; ++j) { float d = v[j] - mean; vs += d * d; }
#pragma unroll
    for (int off = 16; off > 0; off >>= 1) vs += __shfl_xor(vs, off, 32);
    const float rstd = rsqrtf(vs * (1.f / 192.f) + 1e-5f);
    _Float16* po = out + tok * 192;
#pragma unroll
    for (int j = 0; j < 6; ++j) {
        int c = lane + j * 32;
        po[c] = (_Float16)((v[j] - mean) * rstd * g[c] + b[c]);
    }
}

// ---------------------------------------------------------------------------
// WMMA GEMM, block tile 128M x 64N. B^T panel [64 x K] is DMA'd into LDS by the
// Tensor Data Mover (one tensor_load_to_lds from wave 0, waited with
// s_wait_tensorcnt). Each wave owns 32Mx64N: 2 A fragments x 4 B fragments =
// 8 WMMAs per K-step. Requires M%128==0, N%64==0, K%32==0.
// ---------------------------------------------------------------------------
template <bool OUT_F16, bool RESID>
__global__ __launch_bounds__(128) void wmma_gemm_kernel(const _Float16* __restrict__ A,
                                                        const _Float16* __restrict__ BT,
                                                        const float* __restrict__ bias,
                                                        const float* __restrict__ resid,
                                                        _Float16* __restrict__ outH,
                                                        float* __restrict__ outF,
                                                        int M, int N, int K) {
    __shared__ _Float16 Bs[64 * 384];
    const int lane = threadIdx.x & 31;
    const int wave = threadIdx.x >> 5;
    const int col0 = blockIdx.x * 64;
    const int row0 = blockIdx.y * 128 + wave * 32;

    // ---- TDM: DMA B^T panel [64 rows x K halves] -> LDS (ISA ch.8 D#) ----
    if (wave == 0) {
        const unsigned lds_addr = (unsigned)(uintptr_t)(void*)&Bs[0];
        const unsigned long long ga =
            (unsigned long long)(uintptr_t)(BT + (size_t)col0 * K);
        u4v g0;
        g0[0] = 1u;                                        // count=1 (valid, not restore)
        g0[1] = lds_addr;                                  // lds_addr (bytes)
        g0[2] = (unsigned)(ga & 0xffffffffu);              // global_addr[31:0]
        g0[3] = (unsigned)((ga >> 32) & 0x01ffffffu)       // global_addr[56:32]
              | (2u << 30);                                // type=2 ("image")
        u8v g1;
        g1[0] = (1u << 16);                                // wg_mask=0, data_size=1 (2B)
        g1[1] = ((unsigned)K & 0xffffu) << 16;             // tensor_dim0[15:0]
        g1[2] = (((unsigned)K >> 16) & 0xffffu)            // tensor_dim0[31:16]
              | (64u << 16);                               // tensor_dim1[15:0] = 64
        g1[3] = ((unsigned)K << 16);                       // tensor_dim1[31:16]=0 | tile_dim0=K
        g1[4] = 64u;                                       // tile_dim1=64, tile_dim2=0
        g1[5] = (unsigned)K;                               // tensor_dim0_stride[31:0] = K
        g1[6] = 0u;                                        // stride0[47:32]=0, stride1 lo=0
        g1[7] = 0u;                                        // stride1 hi=0
        u4v g2 = (u4v)(0u);
        u4v g3 = (u4v)(0u);
        asm volatile("tensor_load_to_lds %0, %1, %2, %3"
                     :: "s"(g0), "s"(g1), "s"(g2), "s"(g3)
                     : "memory");
        __builtin_amdgcn_s_wait_tensorcnt(0);
    }
    __syncthreads();

    const int mrow = lane & 15;
    const int ncol = lane & 15;
    const int kb   = (lane < 16) ? 0 : 8;   // 16-bit A/B operand layout (ISA 7.12.2)

    v8f acc[2][4] = {};
    const _Float16* pa0 = A + (size_t)(row0 + mrow) * K;
    const _Float16* pa1 = A + (size_t)(row0 + 16 + mrow) * K;
    for (int k0 = 0; k0 < K; k0 += 32) {
        if (k0 + 32 < K) {
            __builtin_prefetch(pa0 + k0 + 32, 0, 1);   // global_prefetch_b8
            __builtin_prefetch(pa1 + k0 + 32, 0, 1);
        }
        const v16h a0 = load_frag(pa0 + k0 + kb);
        const v16h a1 = load_frag(pa1 + k0 + kb);
#pragma unroll
        for (int jt = 0; jt < 4; ++jt) {
            const v16h b = load_frag(&Bs[(jt * 16 + ncol) * K + k0 + kb]);
            acc[0][jt] = __builtin_amdgcn_wmma_f32_16x16x32_f16(false, a0, false, b,
                                                                (short)0, acc[0][jt], false, false);
            acc[1][jt] = __builtin_amdgcn_wmma_f32_16x16x32_f16(false, a1, false, b,
                                                                (short)0, acc[1][jt], false, false);
        }
    }

    const int roff = (lane < 16) ? 0 : 8;
#pragma unroll
    for (int mt = 0; mt < 2; ++mt) {
#pragma unroll
        for (int jt = 0; jt < 4; ++jt) {
            const int c = col0 + jt * 16 + ncol;
            const float bc = bias[c];
#pragma unroll
            for (int v = 0; v < 8; ++v) {
                const size_t r = (size_t)(row0 + mt * 16 + v + roff);
                float val = acc[mt][jt][v] + bc;
                if (RESID) val += resid[r * N + c];
                if (OUT_F16) outH[r * N + c] = (_Float16)val;
                else         outF[r * N + c] = val;
            }
        }
    }
}

// ---------------------------------------------------------------------------
// Window attention: one wave per (window, head), 2 waves / block (40KB LDS).
// V staged transposed in LDS; P written back in-place as f16 over the S rows,
// so both WMMA operands of P*V are vector ds_load_b128.
// qkv: (B*L) x 576 f16 in token order; out: (B*L) x 192 f16 in token order.
// ---------------------------------------------------------------------------
__global__ __launch_bounds__(64) void attn_kernel(const _Float16* __restrict__ qkv,
                                                  const float* __restrict__ rpb,  // 343 x 6
                                                  _Float16* __restrict__ outH,
                                                  int nWperB, int nWH, int nWW,
                                                  int Hh, int Ww, long L) {
    __shared__ float    S[2][64][64];      // per-wave scores; rows become f16 P in-place
    __shared__ _Float16 Vt[2][32][64];     // per-wave V^T: [hd][token]
    const int lane = threadIdx.x & 31;
    const int wave = threadIdx.x >> 5;
    const int task = blockIdx.x * 2 + wave;   // grid sized exactly: total % 2 == 0

    const int head = task % 6;
    const int wid  = task / 6;
    const int b    = wid / nWperB;
    int wrem       = wid % nWperB;
    const int dblk = wrem / nWH; wrem %= nWH;
    const int hblk = wrem / nWW;
    const int wblk = wrem % nWW;

    auto tokidx = [&](int t) -> size_t {
        const int d = dblk * 4 + (t >> 4);
        const int h = hblk * 4 + ((t >> 2) & 3);
        const int w = wblk * 4 + (t & 3);
        return (size_t)b * L + ((size_t)d * Hh + h) * Ww + w;
    };

    const int mrow = lane & 15;
    const int kb   = (lane < 16) ? 0 : 8;
    const int roff = (lane < 16) ? 0 : 8;
    const float scale = 0.17677669529663687f;   // 1/sqrt(32)

    // ---- stage V^T into LDS: each lane handles tokens 2*lane, 2*lane+1 ----
    {
        const int t0 = lane * 2;
        const _Float16* p0 = qkv + tokidx(t0)     * 576 + 384 + head * 32;
        const _Float16* p1 = qkv + tokidx(t0 + 1) * 576 + 384 + head * 32;
        v8h qa[4], qb[4];
#pragma unroll
        for (int c = 0; c < 4; ++c) {
            qa[c] = *(const v8h*)(p0 + c * 8);
            qb[c] = *(const v8h*)(p1 + c * 8);
        }
#pragma unroll
        for (int hd = 0; hd < 32; ++hd) {
            v2h p;
            p[0] = qa[hd >> 3][hd & 7];
            p[1] = qb[hd >> 3][hd & 7];
            *(v2h*)&Vt[wave][hd][t0] = p;
        }
    }

    // ---- S = scale * Q K^T + bias  (16 WMMAs into per-wave LDS slab) ----
    for (int i = 0; i < 4; ++i) {
        const size_t ta = tokidx(i * 16 + mrow);
        const v16h a = load_frag(qkv + ta * 576 + head * 32 + kb);

        for (int j = 0; j < 4; ++j) {
            const int m = j * 16 + (lane & 15);
            const size_t tb = tokidx(m);
            const v16h bm = load_frag(qkv + tb * 576 + 192 + head * 32 + kb);
            v8f c = {};
            c = __builtin_amdgcn_wmma_f32_16x16x32_f16(false, a, false, bm,
                                                       (short)0, c, false, false);
            const int mz = m >> 4, my = (m >> 2) & 3, mx = m & 3;
#pragma unroll
            for (int v = 0; v < 8; ++v) {
                const int n  = i * 16 + v + roff;
                const int nz = n >> 4, ny = (n >> 2) & 3, nx = n & 3;
                const int idx = (nz - mz + 3) * 49 + (ny - my + 3) * 7 + (nx - mx + 3);
                S[wave][n][m] = c[v] * scale + rpb[idx * 6 + head];
            }
        }
    }

    // ---- softmax rows; final pass rewrites each row in place as f16 P ----
#pragma unroll
    for (int rr = 0; rr < 2; ++rr) {
        const int r = lane + rr * 32;
        float mx = -3.4e38f;
        for (int c = 0; c < 64; ++c) mx = fmaxf(mx, S[wave][r][c]);
        float sum = 0.f;
        for (int c = 0; c < 64; ++c) {
            const float e = __expf(S[wave][r][c] - mx);
            S[wave][r][c] = e;
            sum += e;
        }
        const float inv = 1.f / sum;
        _Float16* pr = (_Float16*)&S[wave][r][0];
        for (int c = 0; c < 64; ++c) {            // write offset 2c trails read offset 4c
            const float e = S[wave][r][c];
            pr[c] = (_Float16)(e * inv);
        }
    }

    // ---- O = P V  (16 WMMAs; both operands vector LDS reads) ----
    for (int i = 0; i < 4; ++i) {
        const _Float16* pArow = (const _Float16*)&S[wave][i * 16 + mrow][0];
        v16h a01[2];
#pragma unroll
        for (int kk = 0; kk < 2; ++kk) a01[kk] = load_frag(pArow + kk * 32 + kb);
#pragma unroll
        for (int jn = 0; jn < 2; ++jn) {
            const int n = jn * 16 + (lane & 15);
            v8f c = {};
#pragma unroll
            for (int kk = 0; kk < 2; ++kk) {
                const v16h bm = load_frag(&Vt[wave][n][kk * 32 + kb]);
                c = __builtin_amdgcn_wmma_f32_16x16x32_f16(false, a01[kk], false, bm,
                                                           (short)0, c, false, false);
            }
#pragma unroll
            for (int v = 0; v < 8; ++v) {
                const size_t to = tokidx(i * 16 + v + roff);
                outH[to * 192 + head * 32 + n] = (_Float16)c[v];
            }
        }
    }
}

// ---------------------------------------------------------------------------
// Depthwise 3x3x3 conv (SAME) + BatchNorm + exact GELU. f16 in (token,HID), f16 out.
// ---------------------------------------------------------------------------
__global__ void dwconv_bn_gelu_kernel(const _Float16* __restrict__ h1,
                                      const float* __restrict__ w,   // HID x 27
                                      const float* __restrict__ wb,
                                      const float* __restrict__ bng,
                                      const float* __restrict__ bnb,
                                      const float* __restrict__ bnm,
                                      const float* __restrict__ bnv,
                                      _Float16* __restrict__ out,
                                      int Dd, int Hh, int Ww, long L, long total) {
    const long gid = (long)blockIdx.x * blockDim.x + threadIdx.x;
    if (gid >= total) return;
    const int  c  = (int)(gid % 384);
    const long t  = gid / 384;
    const long li = t % L;
    const long b  = t / L;
    const int wx = (int)(li % Ww);
    const int hy = (int)((li / Ww) % Hh);
    const int dz = (int)(li / ((long)Ww * Hh));

    float acc = 0.f;
#pragma unroll
    for (int kz = 0; kz < 3; ++kz) {
        const int d = dz + kz - 1;
        if (d < 0 || d >= Dd) continue;
#pragma unroll
        for (int ky = 0; ky < 3; ++ky) {
            const int h = hy + ky - 1;
            if (h < 0 || h >= Hh) continue;
#pragma unroll
            for (int kx = 0; kx < 3; ++kx) {
                const int x = wx + kx - 1;
                if (x < 0 || x >= Ww) continue;
                const size_t tt = (size_t)b * L + ((size_t)d * Hh + h) * Ww + x;
                acc += (float)h1[tt * 384 + c] * w[c * 27 + (kz * 3 + ky) * 3 + kx];
            }
        }
    }
    acc += wb[c];
    acc = (acc - bnm[c]) * rsqrtf(bnv[c] + 1e-5f) * bng[c] + bnb[c];
    const float gel = 0.5f * acc * (1.f + erff(acc * 0.70710678118654752f));
    out[gid] = (_Float16)gel;
}

// ---------------------------------------------------------------------------
extern "C" void kernel_launch(void* const* d_in, const int* in_sizes, int n_in,
                              void* d_out, int out_size, void* d_ws, size_t ws_size,
                              hipStream_t stream) {
    const float* x      = (const float*)d_in[0];
    const float* ln1_g  = (const float*)d_in[4];
    const float* ln1_b  = (const float*)d_in[5];
    const float* ln2_g  = (const float*)d_in[6];
    const float* ln2_b  = (const float*)d_in[7];
    const float* qkv_w  = (const float*)d_in[8];
    const float* qkv_b  = (const float*)d_in[9];
    const float* rpb    = (const float*)d_in[10];
    const float* proj_w = (const float*)d_in[11];
    const float* proj_b = (const float*)d_in[12];
    const float* fc1_w  = (const float*)d_in[13];
    const float* fc1_b  = (const float*)d_in[14];
    const float* dw_w   = (const float*)d_in[15];
    const float* dw_b   = (const float*)d_in[16];
    const float* bn_g   = (const float*)d_in[17];
    const float* bn_b   = (const float*)d_in[18];
    const float* bn_m   = (const float*)d_in[19];
    const float* bn_v   = (const float*)d_in[20];
    const float* fc2_w  = (const float*)d_in[21];
    const float* fc2_b  = (const float*)d_in[22];

    const int Dd = 48, Hh = 48, Ww = 48, C = 192, HID = 384;
    const long L  = (long)Dd * Hh * Ww;                       // 110592
    const int  Bb = in_sizes[0] / (int)(L * C);               // 2
    const long NT = (long)Bb * L;                             // 221184 tokens

    // ---- workspace carve-out (~490 MB) ----
    char* ws = (char*)d_ws;
    size_t off = 0;
    auto alloc = [&](size_t bytes) -> char* {
        char* p = ws + off;
        off = (off + bytes + 255) & ~(size_t)255;
        return p;
    };
    _Float16* qkv_wT  = (_Float16*)alloc((size_t)576 * 192 * 2);
    _Float16* proj_wT = (_Float16*)alloc((size_t)192 * 192 * 2);
    _Float16* fc1_wT  = (_Float16*)alloc((size_t)384 * 192 * 2);
    _Float16* fc2_wT  = (_Float16*)alloc((size_t)192 * 384 * 2);
    _Float16* xln_h   = (_Float16*)alloc((size_t)NT * C * 2);    // reused as attn_out
    _Float16* qkv_h   = (_Float16*)alloc((size_t)NT * 576 * 2);  // reused as fc1 out (h1)
    _Float16* y_h     = (_Float16*)alloc((size_t)NT * HID * 2);  // LN2 out; reused as h2
    _Float16* attn_h  = xln_h;
    _Float16* h1_h    = qkv_h;
    _Float16* h2_h    = y_h;
    float*    x1      = (float*)d_out;                            // residual x lives in d_out

    // ---- 1. weight conversion (f32 [K][N] -> f16 [N][K]) ----
    wcvt_kernel<<<(192 * 576 + 255) / 256, 256, 0, stream>>>(qkv_w,  qkv_wT, 192, 576);
    wcvt_kernel<<<(192 * 192 + 255) / 256, 256, 0, stream>>>(proj_w, proj_wT, 192, 192);
    wcvt_kernel<<<(192 * 384 + 255) / 256, 256, 0, stream>>>(fc1_w,  fc1_wT, 192, 384);
    wcvt_kernel<<<(384 * 192 + 255) / 256, 256, 0, stream>>>(fc2_w,  fc2_wT, 384, 192);

    // ---- 2. LN1 ----
    ln_kernel<<<(int)(NT / 8), 256, 0, stream>>>(x, ln1_g, ln1_b, xln_h, NT);

    // ---- 3. QKV GEMM: (NT x 192) * (192 x 576) -> f16 ----
    {
        dim3 grid(576 / 64, (unsigned)(NT / 128));
        wmma_gemm_kernel<true, false><<<grid, 128, 0, stream>>>(
            xln_h, qkv_wT, qkv_b, nullptr, qkv_h, nullptr, (int)NT, 576, 192);
    }

    // ---- 4. window attention ----
    {
        const int nWW = Ww / 4, nWH = (Hh / 4) * nWW, nWperB = (Dd / 4) * nWH;
        const int total = Bb * nWperB * 6;   // 20736, divisible by 2
        attn_kernel<<<total / 2, 64, 0, stream>>>(qkv_h, rpb, attn_h,
                                                  nWperB, nWH, nWW, Hh, Ww, L);
    }

    // ---- 5. proj GEMM + residual -> x1 (f32, in d_out) ----
    {
        dim3 grid(192 / 64, (unsigned)(NT / 128));
        wmma_gemm_kernel<false, true><<<grid, 128, 0, stream>>>(
            attn_h, proj_wT, proj_b, x, nullptr, x1, (int)NT, 192, 192);
    }

    // ---- 6. LN2 ----
    ln_kernel<<<(int)(NT / 8), 256, 0, stream>>>(x1, ln2_g, ln2_b, y_h, NT);

    // ---- 7. fc1 GEMM: (NT x 192) * (192 x 384) -> f16 ----
    {
        dim3 grid(384 / 64, (unsigned)(NT / 128));
        wmma_gemm_kernel<true, false><<<grid, 128, 0, stream>>>(
            y_h, fc1_wT, fc1_b, nullptr, h1_h, nullptr, (int)NT, 384, 192);
    }

    // ---- 8. depthwise conv + BN + GELU -> f16 (overwrites y region) ----
    {
        const long total = NT * HID;
        dwconv_bn_gelu_kernel<<<(int)((total + 255) / 256), 256, 0, stream>>>(
            h1_h, dw_w, dw_b, bn_g, bn_b, bn_m, bn_v, h2_h, Dd, Hh, Ww, L, total);
    }

    // ---- 9. fc2 GEMM + residual (resid and out both d_out; 1:1 element map) ----
    {
        dim3 grid(192 / 64, (unsigned)(NT / 128));
        wmma_gemm_kernel<false, true><<<grid, 128, 0, stream>>>(
            h2_h, fc2_wT, fc2_b, x1, nullptr, (float*)d_out, (int)NT, 192, 384);
    }
}
